// RecurrentModel_6296422056036
// MI455X (gfx1250) — matrix operand order repaired and checked by reading, not compiled, not run on hardware
//
#include <hip/hip_runtime.h>
#include <hip/hip_bf16.h>
#include <math.h>

// ---------------------------------------------------------------------------
// RecurrentModel for MI455X (gfx1250, wave32, WMMA + TDM).
// Pipeline:
//   1) convert x + all weights to bf16 in workspace (one-time per call)
//   2) encoder GEMM   (WMMA bf16)   xe  = x @ enc_w^T + enc_b
//   3) igates GEMM    (WMMA bf16)   ig  = xe @ w_ih^T + b_ih
//   4) persistent GRU scan kernel: w_hh slices DMA'd into LDS by the Tensor
//      Data Mover (196KB/WGP, resident for all 8192 steps), double-buffered
//      h in L2, grid-wide sense barrier per step
//   5) decoder MLP: two WMMA GEMMs (+relu) and a final masked 49-col GEMM
// ---------------------------------------------------------------------------

#define T_STEPS 8192
#define D_INPUT 64
#define HDIM    2048
#define H3      (3 * HDIM)
#define OUT_DIM 49
#define OUT_PAD 64

#define SCAN_WGS   128
#define SLICE      (HDIM / SCAN_WGS)   // 16 hidden units per workgroup
#define SCAN_ROWS  (3 * SLICE)         // 48 rows of w_hh per workgroup

typedef __attribute__((ext_vector_type(16))) __bf16 v16bf;
typedef __attribute__((ext_vector_type(8)))  __bf16 v8bf;
typedef __attribute__((ext_vector_type(8)))  float  v8f;
typedef __attribute__((ext_vector_type(4)))  unsigned int v4u;
typedef __attribute__((ext_vector_type(8)))  int v8i;
typedef __attribute__((ext_vector_type(4)))  int v4i;

#if __has_builtin(__builtin_amdgcn_tensor_load_to_lds) && \
    __has_builtin(__builtin_amdgcn_s_wait_tensorcnt)
#define USE_TDM 1
#else
#define USE_TDM 0
#endif

// ---------------------------------------------------------------------------
// f32 -> bf16 conversion with optional zero-padding (for the 49->64 row pad
// of mlp_w2 so WMMA tiles never read OOB).
// ---------------------------------------------------------------------------
__global__ void cvt_bf16_kernel(const float* __restrict__ src,
                                __bf16* __restrict__ dst,
                                long n, long npad) {
  long stride = (long)gridDim.x * blockDim.x;
  for (long i = (long)blockIdx.x * blockDim.x + threadIdx.x; i < npad; i += stride)
    dst[i] = (i < n) ? (__bf16)src[i] : (__bf16)0.0f;
}

__global__ void init_state_kernel(float* h_bufs, int* bar) {
  int i = blockIdx.x * blockDim.x + threadIdx.x;
  if (i < 2 * HDIM) h_bufs[i] = 0.0f;
  if (i < 8) bar[i] = 0;
}

// ---------------------------------------------------------------------------
// Generic bf16 WMMA GEMM:  C[M,N] = A[M,K] @ W[N,K]^T + bias   (row-major)
// Workgroup = 8 waves = 128(M) x 64(N) tile; each wave does 16x64 via four
// 16x16x32 v_wmma_f32_16x16x32_bf16 per K-step, f32 accumulate.
// A fragment per-lane layout (16-bit A 16x32): lane L holds row (L&15),
// 16 contiguous K at (L>>4)*16.
// B fragment (32x16): lane L holds col (L&15), 16 contiguous K at (L>>4)*16.
// ---------------------------------------------------------------------------
template <bool RELU, bool OUT_BF16>
__global__ void gemm_bf16_wmma(const __bf16* __restrict__ A,
                               const __bf16* __restrict__ W,
                               const float*  __restrict__ bias,
                               void* __restrict__ Cout,
                               int M, int N, int K,
                               int ldc, int nstore, int nbias) {
  const int lane = threadIdx.x & 31;
  const int wv   = threadIdx.x >> 5;               // 0..7
  const int m0   = blockIdx.x * 128 + wv * 16;
  const int n0   = blockIdx.y * 64;
  const int r    = lane & 15;
  const int hi   = lane >> 4;

  const __bf16* arow = A + (size_t)(m0 + r) * K + hi * 16;

  v8f acc[4] = {};

  for (int k0 = 0; k0 < K; k0 += 32) {
    v8bf alo = *(const v8bf*)(arow + k0);
    v8bf ahi = *(const v8bf*)(arow + k0 + 8);
    v16bf a;
#pragma unroll
    for (int i = 0; i < 8; ++i) { a[i] = alo[i]; a[8 + i] = ahi[i]; }

#pragma unroll
    for (int nt = 0; nt < 4; ++nt) {
      const __bf16* wrow = W + (size_t)(n0 + nt * 16 + r) * K + k0 + hi * 16;
      v16bf b = *(const v16bf*)wrow;
      acc[nt] = __builtin_amdgcn_wmma_f32_16x16x32_bf16(
          /*neg_a=*/false, a, /*neg_b=*/false, b,
          /*c_mod=*/(short)0, acc[nt],
          /*reuse_a=*/false, /*reuse_b=*/false);
    }
  }

  // Epilogue: C/D 16x16 f32 layout: VGPR v, lanes 0-15 -> M=v, N=lane;
  // lanes 16-31 -> M=v+8, N=lane-16.
#pragma unroll
  for (int nt = 0; nt < 4; ++nt) {
    int col = n0 + nt * 16 + r;
    float bv = (col < nbias) ? bias[col] : 0.0f;
#pragma unroll
    for (int v = 0; v < 8; ++v) {
      int row = m0 + v + hi * 8;
      float val = acc[nt][v] + bv;
      if (RELU) val = fmaxf(val, 0.0f);
      if (col < nstore) {
        if (OUT_BF16)
          ((__bf16*)Cout)[(size_t)row * ldc + col] = (__bf16)val;
        else
          ((float*)Cout)[(size_t)row * ldc + col] = val;
      }
    }
  }
}

// ---------------------------------------------------------------------------
// Grid-wide sense-reversing barrier (persistent-kernel style).
// bar[0] = arrival counter, bar[1] = generation.
// ---------------------------------------------------------------------------
__device__ __forceinline__ void grid_barrier(int* bar, int nwg) {
  __syncthreads();
  if (threadIdx.x == 0) {
    __threadfence();
    volatile int* gen = bar + 1;
    int g = *gen;
    if (atomicAdd(bar, 1) == nwg - 1) {
      bar[0] = 0;
      __threadfence();
      atomicAdd((int*)gen, 1);
    } else {
      while (*gen == g) __builtin_amdgcn_s_sleep(1);
    }
    __threadfence();
  }
  __syncthreads();
}

// ---------------------------------------------------------------------------
// Persistent GRU scan. Each of 128 WGs owns SLICE=16 hidden units, i.e. the
// 48 rows {i, H+i, 2H+i} of w_hh, held in LDS (48*2048 bf16 = 196 KB).
// The LDS fill is done by the Tensor Data Mover: three 2D-tile descriptors
// (16 rows x 2048 cols, row stride 2048) issued by wave 0 and completed with
// s_wait_tensorcnt 0.
// Per step: stage h (old) into LDS, 8 waves x 6 rows of 2048-long dots with
// 32-lane shuffle reduction, gate math on 16 lanes, write h_new to the other
// h buffer, grid barrier. igates row for t+1 prefetched each step.
// ---------------------------------------------------------------------------
__global__ void gru_scan_kernel(const __bf16* __restrict__ w_hh_b,  // [3H,H]
                                const __bf16* __restrict__ ig_b,    // [T,3H]
                                const float*  __restrict__ b_n,     // [H]
                                float* __restrict__ h_bufs,         // [2,H]
                                __bf16* __restrict__ hs_b,          // [T,H]
                                int* __restrict__ bar) {
  extern __shared__ char smem[];
  __bf16* w_lds  = (__bf16*)smem;                                   // 48*2048
  float*  h_lds  = (float*)(smem + SCAN_ROWS * HDIM * 2);           // 2048
  float*  hg_lds = (float*)(smem + SCAN_ROWS * HDIM * 2 + HDIM * 4); // 48

  const int tid  = threadIdx.x;
  const int wave = tid >> 5;
  const int lane = tid & 31;
  const int wg   = blockIdx.x;
  const int i0   = wg * SLICE;
  const int nwg  = gridDim.x;

#if USE_TDM
  // TDM fill: one D# per gate block g, tile = SLICE rows x HDIM bf16 from
  // w_hh rows [g*H + i0, g*H + i0 + SLICE), destination LDS offset
  // g*SLICE*HDIM*2 from dynamic-LDS base.
  if (tid < 32) {
    const unsigned int lds_base = __builtin_amdgcn_groupstaticsize();
#pragma unroll
    for (int g = 0; g < 3; ++g) {
      unsigned long long ga =
          (unsigned long long)(size_t)(w_hh_b + (size_t)(g * HDIM + i0) * HDIM);
      unsigned int lds_addr = lds_base + (unsigned int)(g * SLICE * HDIM * 2);

      v4u g0;
      g0.x = 1u;                                   // count=1, user descriptor
      g0.y = lds_addr;                             // lds_addr  [63:32]
      g0.z = (unsigned int)(ga & 0xFFFFFFFFull);   // global_addr[95:64]
      g0.w = (unsigned int)((ga >> 32) & 0x1FFFFFFull)
             | (2u << 30);                         // addr[120:96] | type=2

      v8i g1;
      g1[0] = 0x00010000;                          // wg_mask=0, data_size=2B
      g1[1] = (int)(((unsigned)HDIM & 0xFFFFu) << 16);   // tensor_dim0 lo16
      g1[2] = (int)(((unsigned)H3 & 0xFFFFu) << 16);     // dim0 hi=0 | tensor_dim1 lo16
      g1[3] = (int)((unsigned)HDIM << 16);         // dim1 hi=0 | tile_dim0
      g1[4] = SLICE;                               // tile_dim1 | tile_dim2=0
      g1[5] = HDIM;                                // tensor_dim0_stride lo32
      g1[6] = 0;                                   // stride hi | dim1_stride lo
      g1[7] = 0;
      v4i z4 = {0, 0, 0, 0};
      v8i z8 = {0, 0, 0, 0, 0, 0, 0, 0};
      // 6-arg toolchain variant: (g0, g1, g2, g3, extra, cpol)
      __builtin_amdgcn_tensor_load_to_lds(g0, g1, z4, z4, z8, 0);
    }
    __builtin_amdgcn_s_wait_tensorcnt(0);
  }
#else
  // Fallback: manual staged copy (local row g*SLICE+i <- g*H+i0+i).
  for (int lr = 0; lr < SCAN_ROWS; ++lr) {
    int g = lr / SLICE, i = lr % SLICE;
    const __bf16* src = w_hh_b + (size_t)(g * HDIM + i0 + i) * HDIM;
    __bf16* dst = w_lds + lr * HDIM;
    int j = tid * 8;                      // 256 threads * 8 bf16 = one row
    *(uint4*)(dst + j) = *(const uint4*)(src + j);
  }
#endif
  __syncthreads();

  for (int t = 0; t < T_STEPS; ++t) {
    const float* hin  = h_bufs + (t & 1) * HDIM;
    float*       hout = h_bufs + ((t + 1) & 1) * HDIM;

    // Stage old h into LDS (f32, 8 KB).
    for (int j = tid; j < HDIM; j += 256) h_lds[j] = hin[j];
    __syncthreads();

    // Prefetch next step's three igate chunks for this slice.
    if (tid < 3 && (t + 1) < T_STEPS)
      __builtin_prefetch(ig_b + (size_t)(t + 1) * H3 + tid * HDIM + i0, 0, 0);

    // 48 dot products, 6 per wave, 32 lanes each.
#pragma unroll
    for (int rr = 0; rr < 6; ++rr) {
      int lr = wave * 6 + rr;
      const __bf16* wr = w_lds + lr * HDIM;
      float acc = 0.0f;
#pragma unroll 8
      for (int j = lane; j < HDIM; j += 32)
        acc += (float)wr[j] * h_lds[j];
      for (int off = 16; off > 0; off >>= 1)
        acc += __shfl_down(acc, off, 32);
      if (lane == 0) hg_lds[lr] = acc;
    }
    __syncthreads();

    // Gate math for this WG's 16 hidden units.
    if (tid < SLICE) {
      int i = tid;
      float hgr = hg_lds[i];
      float hgz = hg_lds[SLICE + i];
      float hgn = hg_lds[2 * SLICE + i];
      size_t base = (size_t)t * H3;
      float igr = (float)ig_b[base + 0 * HDIM + i0 + i];
      float igz = (float)ig_b[base + 1 * HDIM + i0 + i];
      float ign = (float)ig_b[base + 2 * HDIM + i0 + i];
      float rg = 1.0f / (1.0f + __expf(-(igr + hgr)));
      float zg = 1.0f / (1.0f + __expf(-(igz + hgz)));
      float ng = tanhf(ign + rg * (hgn + b_n[i0 + i]));
      float ho = h_lds[i0 + i];
      float hn = ng + zg * (ho - ng);
      hout[i0 + i] = hn;
      hs_b[(size_t)t * HDIM + i0 + i] = (__bf16)hn;
    }

    grid_barrier(bar, nwg);
  }
}

// ---------------------------------------------------------------------------
// Host-side orchestration.
// ---------------------------------------------------------------------------
static inline size_t align256(size_t x) { return (x + 255) & ~(size_t)255; }

extern "C" void kernel_launch(void* const* d_in, const int* in_sizes, int n_in,
                              void* d_out, int out_size, void* d_ws, size_t ws_size,
                              hipStream_t stream) {
  (void)in_sizes; (void)n_in; (void)out_size; (void)ws_size;

  const float* x      = (const float*)d_in[0];   // [T,64]
  const float* enc_w  = (const float*)d_in[1];   // [H,64]
  const float* enc_b  = (const float*)d_in[2];   // [H]
  const float* w_ih   = (const float*)d_in[3];   // [3H,H]
  const float* w_hh   = (const float*)d_in[4];   // [3H,H]
  const float* b_ih   = (const float*)d_in[5];   // [3H]
  const float* b_n    = (const float*)d_in[6];   // [H]
  const float* mlp_w0 = (const float*)d_in[7];   // [H,H]
  const float* mlp_b0 = (const float*)d_in[8];   // [H]
  const float* mlp_w1 = (const float*)d_in[9];   // [H,H]
  const float* mlp_b1 = (const float*)d_in[10];  // [H]
  const float* mlp_w2 = (const float*)d_in[11];  // [49,H]
  const float* mlp_b2 = (const float*)d_in[12];  // [49]
  float* out = (float*)d_out;                    // [T,49]

  // Workspace carve-up (all 256B-aligned).
  char* ws = (char*)d_ws;
  size_t off = 0;
  auto carve = [&](size_t bytes) { char* p = ws + off; off += align256(bytes); return p; };

  __bf16* x_b     = (__bf16*)carve((size_t)T_STEPS * D_INPUT * 2);
  __bf16* enc_w_b = (__bf16*)carve((size_t)HDIM * D_INPUT * 2);
  __bf16* w_ih_b  = (__bf16*)carve((size_t)H3 * HDIM * 2);
  __bf16* w_hh_b  = (__bf16*)carve((size_t)H3 * HDIM * 2);
  __bf16* w0_b    = (__bf16*)carve((size_t)HDIM * HDIM * 2);
  __bf16* w1_b    = (__bf16*)carve((size_t)HDIM * HDIM * 2);
  __bf16* w2_b    = (__bf16*)carve((size_t)OUT_PAD * HDIM * 2);
  __bf16* ig_b    = (__bf16*)carve((size_t)T_STEPS * H3 * 2);
  __bf16* xe_b    = (__bf16*)carve((size_t)T_STEPS * HDIM * 2);
  __bf16* hs_b    = (__bf16*)carve((size_t)T_STEPS * HDIM * 2);
  float*  h_bufs  = (float*)carve(2 * HDIM * 4);
  int*    bar     = (int*)carve(256);
  // Alias decoder activations onto ig region (igates dead after the scan).
  __bf16* a1_b = ig_b;
  __bf16* a2_b = ig_b + (size_t)T_STEPS * HDIM;

  // 1) bf16 conversions.
  auto cvt = [&](const float* s, __bf16* d, long n, long npad) {
    int blocks = (int)((npad + 255) / 256);
    if (blocks > 4096) blocks = 4096;
    cvt_bf16_kernel<<<blocks, 256, 0, stream>>>(s, d, n, npad);
  };
  cvt(x, x_b, (long)T_STEPS * D_INPUT, (long)T_STEPS * D_INPUT);
  cvt(enc_w, enc_w_b, (long)HDIM * D_INPUT, (long)HDIM * D_INPUT);
  cvt(w_ih, w_ih_b, (long)H3 * HDIM, (long)H3 * HDIM);
  cvt(w_hh, w_hh_b, (long)H3 * HDIM, (long)H3 * HDIM);
  cvt(mlp_w0, w0_b, (long)HDIM * HDIM, (long)HDIM * HDIM);
  cvt(mlp_w1, w1_b, (long)HDIM * HDIM, (long)HDIM * HDIM);
  cvt(mlp_w2, w2_b, (long)OUT_DIM * HDIM, (long)OUT_PAD * HDIM);  // zero-pad rows 49..63

  init_state_kernel<<<(2 * HDIM + 255) / 256, 256, 0, stream>>>(h_bufs, bar);

  // 2) encoder GEMM: xe = x @ enc_w^T + enc_b   [T,H]
  gemm_bf16_wmma<false, true><<<dim3(T_STEPS / 128, HDIM / 64), 256, 0, stream>>>(
      x_b, enc_w_b, enc_b, xe_b, T_STEPS, HDIM, D_INPUT, HDIM, HDIM, HDIM);

  // 3) igates GEMM: ig = xe @ w_ih^T + b_ih   [T,3H]
  gemm_bf16_wmma<false, true><<<dim3(T_STEPS / 128, H3 / 64), 256, 0, stream>>>(
      xe_b, w_ih_b, b_ih, ig_b, T_STEPS, H3, HDIM, H3, H3, H3);

  // 4) GRU scan (persistent; w_hh resident in LDS via TDM).
  size_t scan_smem = (size_t)SCAN_ROWS * HDIM * 2 + HDIM * 4 + SCAN_ROWS * 4;
  gru_scan_kernel<<<SCAN_WGS, 256, scan_smem, stream>>>(
      w_hh_b, ig_b, b_n, h_bufs, hs_b, bar);

  // 5) decoder MLP.
  gemm_bf16_wmma<true, true><<<dim3(T_STEPS / 128, HDIM / 64), 256, 0, stream>>>(
      hs_b, w0_b, mlp_b0, a1_b, T_STEPS, HDIM, HDIM, HDIM, HDIM, HDIM);
  gemm_bf16_wmma<true, true><<<dim3(T_STEPS / 128, HDIM / 64), 256, 0, stream>>>(
      a1_b, w1_b, mlp_b1, a2_b, T_STEPS, HDIM, HDIM, HDIM, HDIM, HDIM);
  gemm_bf16_wmma<false, false><<<dim3(T_STEPS / 128, OUT_PAD / 64), 256, 0, stream>>>(
      a2_b, w2_b, mlp_b2, out, T_STEPS, OUT_PAD, HDIM, OUT_DIM, OUT_DIM, OUT_DIM);
}
